// TransformerBlock_45397804319143
// MI455X (gfx1250) — matrix operand order repaired and checked by reading, not compiled, not run on hardware
//
#include <hip/hip_runtime.h>
#include <hip/hip_bf16.h>

#define SLEN 2048
#define HID  1024
#define NH   16
#define HD   64
#define DFF  4096
#define NEXP 8
#define EPS  1e-6f

typedef __attribute__((ext_vector_type(16))) _Float16 v16h;
typedef __attribute__((ext_vector_type(8)))  _Float16 v8h;
typedef __attribute__((ext_vector_type(8)))  float    v8f;

// ---------------------------------------------------------------------------
// CDNA5 async global->LDS staging (ASYNCcnt path), with safe fallback.
// Builtin signature (from compiler diagnostic): (int4 as1*, int4 as3*, Ii, Ii)
// ---------------------------------------------------------------------------
#if defined(__AMDGCN__) && __has_builtin(__builtin_amdgcn_global_load_async_to_lds_b128)
#define HAVE_ASYNC_LDS 1
typedef int v4i_vs __attribute__((vector_size(16)));
typedef __attribute__((address_space(1))) v4i_vs* as1_v4i_ptr;
typedef __attribute__((address_space(3))) v4i_vs* as3_v4i_ptr;
#else
#define HAVE_ASYNC_LDS 0
#endif

__device__ __forceinline__ void async_copy16(const void* gsrc, void* ldst) {
#if HAVE_ASYNC_LDS
  as1_v4i_ptr g = (as1_v4i_ptr)(unsigned long long)(uintptr_t)gsrc;
  as3_v4i_ptr l = (as3_v4i_ptr)(unsigned)(uintptr_t)ldst;
  __builtin_amdgcn_global_load_async_to_lds_b128(g, l, 0, 0);
#else
  *(uint4*)ldst = *(const uint4*)gsrc;
#endif
}

__device__ __forceinline__ void async_wait() {
#if HAVE_ASYNC_LDS
#if __has_builtin(__builtin_amdgcn_s_wait_asynccnt)
  __builtin_amdgcn_s_wait_asynccnt(0);
#else
  asm volatile("s_wait_asynccnt 0x0" ::: "memory");
#endif
#endif
}

// ---------------------------------------------------------------------------
// RMSNorm: fp32 row -> f16 normalized row
// ---------------------------------------------------------------------------
__global__ void rmsnorm_f16_kernel(const float* __restrict__ x,
                                   const float* __restrict__ w,
                                   _Float16* __restrict__ out) {
  __shared__ float red[256];
  const int row = blockIdx.x;
  const int tid = threadIdx.x;
  const float* xr = x + (size_t)row * HID;
  float ss = 0.f;
  float vals[4];
  for (int i = 0; i < 4; ++i) {
    float v = xr[tid + i * 256];
    vals[i] = v;
    ss += v * v;
  }
  red[tid] = ss;
  __syncthreads();
  for (int off = 128; off > 0; off >>= 1) {
    if (tid < off) red[tid] += red[tid + off];
    __syncthreads();
  }
  const float scale = rsqrtf(red[0] / (float)HID + EPS);
  _Float16* orow = out + (size_t)row * HID;
  for (int i = 0; i < 4; ++i) {
    const int c = tid + i * 256;
    orow[c] = (_Float16)(vals[i] * w[c] * scale);
  }
}

// ---------------------------------------------------------------------------
// Generic WMMA GEMM:  C[M,N] = A[M,K] (f16) * B[N,K]^T (fp32 weights)
// Block = 256 threads (8 waves), tile 64x64, K-step 32.
// A tile staged via GLOBAL_LOAD_ASYNC_TO_LDS_B128 (ASYNCcnt); B tile staged
// through VGPRs because it needs an fp32->f16 convert.
// Epilogue modes:
//   0 F16OUT : outH = C
//   1 RESID  : outF = C + res
//   2 SWIGLU : gathered-A rows (toklist); outH = silu(C1) * C2   (B2 = w3)
//   3 SCATTER: outF[slot*S*HID + token*N + col] = gate * C
// ---------------------------------------------------------------------------
#define EP_F16OUT  0
#define EP_RESID   1
#define EP_SWIGLU  2
#define EP_SCATTER 3

__global__ void gemm_wmma_kernel(const _Float16* __restrict__ A, int K, int M,
                                 const float* __restrict__ B1,
                                 const float* __restrict__ B2, int N,
                                 const int* __restrict__ count,
                                 const int* __restrict__ toklist,
                                 const float* __restrict__ gatelist,
                                 const float* __restrict__ res,
                                 float* __restrict__ outF,
                                 _Float16* __restrict__ outH,
                                 int mode) {
  __shared__ _Float16 As[64][40];
  __shared__ _Float16 Bs[64][40];
  __shared__ _Float16 Bs2[64][40];

  const int rowbase = blockIdx.y * 64;
  const int colbase = blockIdx.x * 64;
  int Mv = M;
  if (count) {
    Mv = *count;
    if (rowbase >= Mv) return;
  }

  const int tid  = threadIdx.x;
  const int lane = tid & 31;
  const int wave = tid >> 5;
  const int mt = wave & 3;            // 16-row tile within block
  const int ntb = (wave >> 2) * 2;    // first of two 16-col tiles

  const bool gather = (mode == EP_SWIGLU);

  v8f acc0 = {}; v8f acc1 = {};
  v8f acc0b = {}; v8f acc1b = {};

  const int lr = tid >> 2;            // 0..63 row within tile
  const int lc = (tid & 3) * 8;       // 0..24 col base (8 elems)

  // resolve this thread's A source row once (clamped for ragged tails)
  int arow = 0;
  {
    const int gr = rowbase + lr;
    if (gr < Mv) arow = gather ? (toklist[gr] & 0xFFFFF) : gr;
  }

  for (int kk = 0; kk < K; kk += 32) {
    // --- stage A tile: async DMA straight into LDS ---
    {
      const _Float16* src = A + (size_t)arow * K + kk + lc;
      async_copy16(src, &As[lr][lc]);
      __builtin_prefetch(src + 32, 0, 0);
    }
    // --- stage B tile(s): fp32 weights -> f16 via VGPRs ---
    {
      const int n = colbase + lr;
      const float4 b0 = *(const float4*)(B1 + (size_t)n * K + kk + lc);
      const float4 b1 = *(const float4*)(B1 + (size_t)n * K + kk + lc + 4);
      __builtin_prefetch(B1 + (size_t)n * K + kk + 32 + lc, 0, 0);
      Bs[lr][lc + 0] = (_Float16)b0.x; Bs[lr][lc + 1] = (_Float16)b0.y;
      Bs[lr][lc + 2] = (_Float16)b0.z; Bs[lr][lc + 3] = (_Float16)b0.w;
      Bs[lr][lc + 4] = (_Float16)b1.x; Bs[lr][lc + 5] = (_Float16)b1.y;
      Bs[lr][lc + 6] = (_Float16)b1.z; Bs[lr][lc + 7] = (_Float16)b1.w;
      if (mode == EP_SWIGLU) {
        const float4 c0 = *(const float4*)(B2 + (size_t)n * K + kk + lc);
        const float4 c1 = *(const float4*)(B2 + (size_t)n * K + kk + lc + 4);
        Bs2[lr][lc + 0] = (_Float16)c0.x; Bs2[lr][lc + 1] = (_Float16)c0.y;
        Bs2[lr][lc + 2] = (_Float16)c0.z; Bs2[lr][lc + 3] = (_Float16)c0.w;
        Bs2[lr][lc + 4] = (_Float16)c1.x; Bs2[lr][lc + 5] = (_Float16)c1.y;
        Bs2[lr][lc + 6] = (_Float16)c1.z; Bs2[lr][lc + 7] = (_Float16)c1.w;
      }
    }
    async_wait();
    __syncthreads();

    // --- A fragment (contiguous k = 16*(lane>>4) + s, same as B) ---
    const int am = mt * 16 + (lane & 15);
    const int ko = (lane >> 4) * 16;
    v8h alo = *(const v8h*)&As[am][ko];
    v8h ahi = *(const v8h*)&As[am][ko + 8];
    v16h afrag = __builtin_shufflevector(alo, ahi, 0,1,2,3,4,5,6,7,8,9,10,11,12,13,14,15);

    // --- B fragments: lane (l&15) reads 16 contiguous halves of weight row ---
    const int kb = (lane >> 4) * 16;
    {
      const int bn = ntb * 16 + (lane & 15);
      v8h b0 = *(const v8h*)&Bs[bn][kb];
      v8h b1 = *(const v8h*)&Bs[bn][kb + 8];
      v16h bf = __builtin_shufflevector(b0, b1, 0,1,2,3,4,5,6,7,8,9,10,11,12,13,14,15);
      acc0 = __builtin_amdgcn_wmma_f32_16x16x32_f16(false, afrag, false, bf,
                                                    (short)0, acc0, false, false);
      if (mode == EP_SWIGLU) {
        v8h c0 = *(const v8h*)&Bs2[bn][kb];
        v8h c1 = *(const v8h*)&Bs2[bn][kb + 8];
        v16h cf = __builtin_shufflevector(c0, c1, 0,1,2,3,4,5,6,7,8,9,10,11,12,13,14,15);
        acc0b = __builtin_amdgcn_wmma_f32_16x16x32_f16(false, afrag, false, cf,
                                                       (short)0, acc0b, false, false);
      }
    }
    {
      const int bn = (ntb + 1) * 16 + (lane & 15);
      v8h b0 = *(const v8h*)&Bs[bn][kb];
      v8h b1 = *(const v8h*)&Bs[bn][kb + 8];
      v16h bf = __builtin_shufflevector(b0, b1, 0,1,2,3,4,5,6,7,8,9,10,11,12,13,14,15);
      acc1 = __builtin_amdgcn_wmma_f32_16x16x32_f16(false, afrag, false, bf,
                                                    (short)0, acc1, false, false);
      if (mode == EP_SWIGLU) {
        v8h c0 = *(const v8h*)&Bs2[bn][kb];
        v8h c1 = *(const v8h*)&Bs2[bn][kb + 8];
        v16h cf = __builtin_shufflevector(c0, c1, 0,1,2,3,4,5,6,7,8,9,10,11,12,13,14,15);
        acc1b = __builtin_amdgcn_wmma_f32_16x16x32_f16(false, afrag, false, cf,
                                                       (short)0, acc1b, false, false);
      }
    }
    __syncthreads();
  }

  // --- epilogue (ISA C/D layout: lane&15 = col, (lane>>4)*8+r = row) ---
  const int nl = lane & 15;
  const int mrb = mt * 16 + (lane >> 4) * 8;
  for (int t = 0; t < 2; ++t) {
    const int gcol = colbase + (ntb + t) * 16 + nl;
    v8f acc  = t ? acc1  : acc0;
    v8f accb = t ? acc1b : acc0b;
    for (int r = 0; r < 8; ++r) {
      const int gr = rowbase + mrb + r;
      if (gr >= Mv) continue;
      const float c = acc[r];
      if (mode == EP_F16OUT) {
        outH[(size_t)gr * N + gcol] = (_Float16)c;
      } else if (mode == EP_RESID) {
        outF[(size_t)gr * N + gcol] = c + res[(size_t)gr * N + gcol];
      } else if (mode == EP_SWIGLU) {
        const float c2 = accb[r];
        const float s = c / (1.f + __expf(-c));
        outH[(size_t)gr * N + gcol] = (_Float16)(s * c2);
      } else { // EP_SCATTER
        const int pk = toklist[gr];
        const int tokn = pk & 0xFFFFF;
        const int slot = pk >> 20;
        outF[(size_t)slot * SLEN * HID + (size_t)tokn * N + gcol] = gatelist[gr] * c;
      }
    }
  }
}

// ---------------------------------------------------------------------------
// In-place RoPE on f16 q and k ([S, HID], head-sliced by 64)
// ---------------------------------------------------------------------------
__global__ void rope_kernel(_Float16* __restrict__ q, _Float16* __restrict__ k,
                            const float* __restrict__ rot) {
  const int idx = blockIdx.x * 256 + threadIdx.x;     // pair index
  const int t = idx >> 9;          // token (512 pairs/token)
  const int p = idx & 511;
  const int head = p >> 5;
  const int i = p & 31;
  const float ang = rot[t * (HD / 2) + i];
  const float cs = __cosf(ang), sn = __sinf(ang);
  const size_t base = (size_t)t * HID + head * HD + 2 * i;
  {
    float t1 = (float)q[base], t2 = (float)q[base + 1];
    q[base] = (_Float16)(t1 * cs - t2 * sn);
    q[base + 1] = (_Float16)(t1 * sn + t2 * cs);
  }
  {
    float t1 = (float)k[base], t2 = (float)k[base + 1];
    k[base] = (_Float16)(t1 * cs - t2 * sn);
    k[base + 1] = (_Float16)(t1 * sn + t2 * cs);
  }
}

// ---------------------------------------------------------------------------
// Sliding-window attention, one wave per (16-query tile, head).
// Keys in [q0-128, q0+15] -> 9 aligned 16-key blocks. QK^T and attn*V in WMMA.
// Q/K tiles staged with async-to-LDS; V transposed through VGPRs.
// ---------------------------------------------------------------------------
__global__ __launch_bounds__(32)
void attn_kernel(const _Float16* __restrict__ qh, const _Float16* __restrict__ kh,
                 const _Float16* __restrict__ vh, _Float16* __restrict__ oh) {
  __shared__ _Float16 qt[16][72];
  __shared__ _Float16 kt[16][72];
  __shared__ float    sc[16][160];
  __shared__ _Float16 at[16][160];
  __shared__ _Float16 vtT[64][168];   // [hd][key] (transposed V)

  const int qtile = blockIdx.x;
  const int head  = blockIdx.y;
  const int q0 = qtile * 16;
  const int lane = threadIdx.x;

  // load Q tile: 16 rows x 64 halves (async DMA to LDS)
  {
    const int r = lane >> 1, cb = (lane & 1) * 32;
    const _Float16* src = qh + (size_t)(q0 + r) * HID + head * HD + cb;
    async_copy16(src,      &qt[r][cb]);
    async_copy16(src + 8,  &qt[r][cb + 8]);
    async_copy16(src + 16, &qt[r][cb + 16]);
    async_copy16(src + 24, &qt[r][cb + 24]);
  }

  const int nl = lane & 15;
  const int mrb = (lane >> 4) * 8;
  const int ko = (lane >> 4) * 16;
  const int kb16 = (lane >> 4) * 16;

  for (int kbi = 0; kbi < 9; ++kbi) {
    const int kblk = qtile - 8 + kbi;
    if (kblk >= 0) {
      // stage K tile (async) and transposed V tile (via VGPRs)
      const int r = lane >> 1, cb = (lane & 1) * 32;
      const _Float16* ks = kh + (size_t)(kblk * 16 + r) * HID + head * HD + cb;
      async_copy16(ks,      &kt[r][cb]);
      async_copy16(ks + 8,  &kt[r][cb + 8]);
      async_copy16(ks + 16, &kt[r][cb + 16]);
      async_copy16(ks + 24, &kt[r][cb + 24]);
      const _Float16* vs = vh + (size_t)(kblk * 16 + r) * HID + head * HD + cb;
      for (int j = 0; j < 32; ++j) vtT[cb + j][kbi * 16 + r] = vs[j];
      async_wait();
      __syncthreads();

      // scores: contract over HD=64 -> two K=32 WMMAs
      v8f s = {};
      for (int kc = 0; kc < 2; ++kc) {
        const int k0 = kc * 32;
        v8h alo = *(const v8h*)&qt[nl][k0 + ko];
        v8h ahi = *(const v8h*)&qt[nl][k0 + ko + 8];
        v16h af = __builtin_shufflevector(alo, ahi, 0,1,2,3,4,5,6,7,8,9,10,11,12,13,14,15);
        v8h b0 = *(const v8h*)&kt[nl][k0 + kb16];
        v8h b1 = *(const v8h*)&kt[nl][k0 + kb16 + 8];
        v16h bf = __builtin_shufflevector(b0, b1, 0,1,2,3,4,5,6,7,8,9,10,11,12,13,14,15);
        s = __builtin_amdgcn_wmma_f32_16x16x32_f16(false, af, false, bf,
                                                   (short)0, s, false, false);
      }
      // mask + scale, spill to LDS
      for (int r2 = 0; r2 < 8; ++r2) {
        const int m = mrb + r2;
        const int gq = q0 + m;
        const int gk = kblk * 16 + nl;
        const int d = gq - gk;
        float v = s[r2] * 0.125f;
        if (d < 0 || d > 128) v = -3.0e38f;
        sc[m][kbi * 16 + nl] = v;
      }
      __syncthreads();
    } else {
      for (int r2 = 0; r2 < 8; ++r2) sc[mrb + r2][kbi * 16 + nl] = -3.0e38f;
      const int r = lane >> 1, cb = (lane & 1) * 32;
      for (int j = 0; j < 32; ++j) vtT[cb + j][kbi * 16 + r] = (_Float16)0.f;
      __syncthreads();
    }
  }

  // softmax over 144 valid key columns; one lane per row
  if (lane < 16) {
    float mx = -3.4e38f;
    for (int c = 0; c < 144; ++c) mx = fmaxf(mx, sc[lane][c]);
    float sum = 0.f;
    for (int c = 0; c < 144; ++c) {
      const float e = __expf(sc[lane][c] - mx);
      sc[lane][c] = e;
      sum += e;
    }
    const float inv = 1.f / sum;
    for (int c = 0; c < 144; ++c) at[lane][c] = (_Float16)(sc[lane][c] * inv);
    for (int c = 144; c < 160; ++c) at[lane][c] = (_Float16)0.f;
  }
  __syncthreads();

  // out = attn(16x160) @ V(160x64): 4 col tiles x 5 K-chunks of 32 keys
  for (int nt = 0; nt < 4; ++nt) {
    v8f o = {};
    for (int kc = 0; kc < 5; ++kc) {
      const int k0 = kc * 32;
      v8h alo = *(const v8h*)&at[nl][k0 + ko];
      v8h ahi = *(const v8h*)&at[nl][k0 + ko + 8];
      v16h af = __builtin_shufflevector(alo, ahi, 0,1,2,3,4,5,6,7,8,9,10,11,12,13,14,15);
      const int bn = nt * 16 + nl;
      v8h b0 = *(const v8h*)&vtT[bn][k0 + kb16];
      v8h b1 = *(const v8h*)&vtT[bn][k0 + kb16 + 8];
      v16h bf = __builtin_shufflevector(b0, b1, 0,1,2,3,4,5,6,7,8,9,10,11,12,13,14,15);
      o = __builtin_amdgcn_wmma_f32_16x16x32_f16(false, af, false, bf,
                                                 (short)0, o, false, false);
    }
    for (int r2 = 0; r2 < 8; ++r2) {
      const int m = mrb + r2;
      oh[(size_t)(q0 + m) * HID + head * HD + nt * 16 + nl] = (_Float16)o[r2];
    }
  }
}

// ---------------------------------------------------------------------------
// Router: logits, top-2, softmax gates, build per-expert token lists.
// One wave per token (8 waves/block).
// ---------------------------------------------------------------------------
__global__ void router_kernel(const _Float16* __restrict__ xn,
                              const float* __restrict__ rw,
                              int* __restrict__ counts,
                              int* __restrict__ tok,
                              float* __restrict__ gate) {
  const int lane = threadIdx.x & 31;
  const int wave = threadIdx.x >> 5;
  const int t = blockIdx.x * 8 + wave;
  float lg[NEXP];
  for (int e = 0; e < NEXP; ++e) {
    float s = 0.f;
    for (int i = lane; i < HID; i += 32)
      s += (float)xn[(size_t)t * HID + i] * rw[(size_t)e * HID + i];
    for (int off = 16; off > 0; off >>= 1) s += __shfl_xor(s, off, 32);
    lg[e] = s;
  }
  if (lane == 0) {
    int i0 = 0;
    for (int e = 1; e < NEXP; ++e) if (lg[e] > lg[i0]) i0 = e;
    int i1 = (i0 == 0) ? 1 : 0;
    for (int e = 0; e < NEXP; ++e)
      if (e != i0 && lg[e] > lg[i1]) i1 = e;
    const float e1 = __expf(lg[i1] - lg[i0]);
    const float inv = 1.f / (1.f + e1);
    const float g0 = inv, g1 = e1 * inv;
    int p0 = atomicAdd(&counts[i0], 1);
    tok[i0 * SLEN + p0] = t;               // slot 0
    gate[i0 * SLEN + p0] = g0;
    int p1 = atomicAdd(&counts[i1], 1);
    tok[i1 * SLEN + p1] = t | (1 << 20);   // slot 1
    gate[i1 * SLEN + p1] = g1;
  }
}

// ---------------------------------------------------------------------------
// Final: out = x1 + slot0 + slot1
// ---------------------------------------------------------------------------
__global__ void final_add_kernel(const float* __restrict__ x1,
                                 const float* __restrict__ slots,
                                 float* __restrict__ out) {
  const size_t i = (size_t)blockIdx.x * 256 + threadIdx.x;
  out[i] = x1[i] + slots[i] + slots[(size_t)SLEN * HID + i];
}

// ---------------------------------------------------------------------------
extern "C" void kernel_launch(void* const* d_in, const int* in_sizes, int n_in,
                              void* d_out, int out_size, void* d_ws, size_t ws_size,
                              hipStream_t stream) {
  const float* x     = (const float*)d_in[0];
  const float* wq    = (const float*)d_in[1];
  const float* wk    = (const float*)d_in[2];
  const float* wv    = (const float*)d_in[3];
  const float* wo    = (const float*)d_in[4];
  const float* n1w   = (const float*)d_in[5];
  const float* n2w   = (const float*)d_in[6];
  const float* rw    = (const float*)d_in[7];
  const float* w1    = (const float*)d_in[8];
  const float* w3    = (const float*)d_in[9];
  const float* w2    = (const float*)d_in[10];
  const float* rot   = (const float*)d_in[11];
  float* out = (float*)d_out;

  // workspace layout (256B-aligned slabs)
  char* ws = (char*)d_ws;
  size_t off = 0;
  auto alloc = [&](size_t bytes) { void* p = ws + off; off = (off + bytes + 255) & ~(size_t)255; return p; };
  _Float16* xn1_h  = (_Float16*)alloc((size_t)SLEN * HID * 2);
  _Float16* q_h    = (_Float16*)alloc((size_t)SLEN * HID * 2);
  _Float16* k_h    = (_Float16*)alloc((size_t)SLEN * HID * 2);
  _Float16* v_h    = (_Float16*)alloc((size_t)SLEN * HID * 2);
  _Float16* attn_h = (_Float16*)alloc((size_t)SLEN * HID * 2);
  float*    x1     = (float*)alloc((size_t)SLEN * HID * 4);
  _Float16* xn2_h  = (_Float16*)alloc((size_t)SLEN * HID * 2);
  int*      counts = (int*)alloc(NEXP * 4);
  int*      tok    = (int*)alloc((size_t)NEXP * SLEN * 4);
  float*    gate   = (float*)alloc((size_t)NEXP * SLEN * 4);
  _Float16* h_h    = (_Float16*)alloc((size_t)SLEN * DFF * 2);
  float*    slots  = (float*)alloc((size_t)2 * SLEN * HID * 4);

  // 1) RMSNorm 1
  rmsnorm_f16_kernel<<<SLEN, 256, 0, stream>>>(x, n1w, xn1_h);

  // 2) Q/K/V projections (WMMA)
  dim3 gProj(HID / 64, SLEN / 64);
  gemm_wmma_kernel<<<gProj, 256, 0, stream>>>(xn1_h, HID, SLEN, wq, nullptr, HID,
      nullptr, nullptr, nullptr, nullptr, nullptr, q_h, EP_F16OUT);
  gemm_wmma_kernel<<<gProj, 256, 0, stream>>>(xn1_h, HID, SLEN, wk, nullptr, HID,
      nullptr, nullptr, nullptr, nullptr, nullptr, k_h, EP_F16OUT);
  gemm_wmma_kernel<<<gProj, 256, 0, stream>>>(xn1_h, HID, SLEN, wv, nullptr, HID,
      nullptr, nullptr, nullptr, nullptr, nullptr, v_h, EP_F16OUT);

  // 3) RoPE in place on q,k
  rope_kernel<<<(SLEN * HID / 2) / 256, 256, 0, stream>>>(q_h, k_h, rot);

  // 4) windowed attention
  attn_kernel<<<dim3(SLEN / 16, NH), 32, 0, stream>>>(q_h, k_h, v_h, attn_h);

  // 5) O projection + residual -> x1
  gemm_wmma_kernel<<<gProj, 256, 0, stream>>>(attn_h, HID, SLEN, wo, nullptr, HID,
      nullptr, nullptr, nullptr, x, x1, nullptr, EP_RESID);

  // 6) RMSNorm 2
  rmsnorm_f16_kernel<<<SLEN, 256, 0, stream>>>(x1, n2w, xn2_h);

  // 7) routing
  (void)hipMemsetAsync(counts, 0, NEXP * 4, stream);
  (void)hipMemsetAsync(slots, 0, (size_t)2 * SLEN * HID * 4, stream);
  router_kernel<<<SLEN / 8, 256, 0, stream>>>(xn2_h, rw, counts, tok, gate);

  // 8) per-expert gathered SwiGLU + scatter-scaled down-proj
  for (int e = 0; e < NEXP; ++e) {
    gemm_wmma_kernel<<<dim3(DFF / 64, SLEN / 64), 256, 0, stream>>>(
        xn2_h, HID, SLEN, w1 + (size_t)e * DFF * HID, w3 + (size_t)e * DFF * HID,
        DFF, counts + e, tok + (size_t)e * SLEN, nullptr, nullptr,
        nullptr, h_h, EP_SWIGLU);
    gemm_wmma_kernel<<<dim3(HID / 64, SLEN / 64), 256, 0, stream>>>(
        h_h, DFF, SLEN, w2 + (size_t)e * HID * DFF, nullptr, HID,
        counts + e, tok + (size_t)e * SLEN, gate + (size_t)e * SLEN, nullptr,
        slots, nullptr, EP_SCATTER);
  }

  // 9) final residual add
  final_add_kernel<<<(SLEN * HID) / 256, 256, 0, stream>>>(x1, slots, out);
}
